// MultiMLP_79800492360086
// MI455X (gfx1250) — matrix operand order repaired and checked by reading, not compiled, not run on hardware
//
#include <hip/hip_runtime.h>

typedef __attribute__((ext_vector_type(16))) __bf16 v16bf;
typedef __attribute__((ext_vector_type(8)))  float  v8f;

#define Bsz 8
#define Psz 1024
#define Nsz 32
#define Lsz 125
#define Hsz 256
#define Esz 128   // 3 + 125
#define MT  64    // points per workgroup

__device__ __forceinline__ unsigned short f2bf(float f) {
    union { float f; unsigned u; } c; c.f = f;
    unsigned x = c.u;
    unsigned r = x + 0x7FFFu + ((x >> 16) & 1u);   // round to nearest even
    return (unsigned short)(r >> 16);
}
__device__ __forceinline__ float bf2f(unsigned short h) {
    union { unsigned u; float f; } c; c.u = ((unsigned)h) << 16;
    return c.f;
}

union Frag16 { unsigned u[8]; uint4 q[2]; v16bf v; };

// ---------------------------------------------------------------------------
// Weight prepack: f32 [N][K][Hsz] -> packed bf16 pairs, fragment-major layout
// Wp[((n*Hsz + col) * (K/2)) + k2] = pack(bf16(W[n][2k2][col]), bf16(W[n][2k2+1][col]))
// A lane's B-fragment (8 dwords, k2 = 16*ks + 8*half + r) is then contiguous.
// ---------------------------------------------------------------------------
__global__ void __launch_bounds__(256) prepack_kernel(
    const float* __restrict__ W, unsigned* __restrict__ Wp, int Khalf, int total)
{
    int idx = blockIdx.x * 256 + threadIdx.x;
    if (idx >= total) return;
    const int k2  = idx % Khalf;
    const int t   = idx / Khalf;
    const int col = t % Hsz;
    const int n   = t / Hsz;
    const float* base = W + (size_t)(n * Khalf * 2 + 2 * k2) * Hsz + col;
    const unsigned lo = f2bf(base[0]);
    const unsigned hi = f2bf(base[Hsz]);
    Wp[idx] = lo | (hi << 16);
}

// ---------------------------------------------------------------------------
// One MLP layer: out[MT][Hsz] = relu(act[MT][K] @ W + b), bf16 in/out, f32 acc.
// Wave w owns columns [w*32, w*32+32): RT row tiles x 2 col tiles of 16x16.
// ---------------------------------------------------------------------------
template<int K, int RT>
__device__ __forceinline__ void gemm_layer(
    const unsigned short* __restrict__ actL,   // LDS [MT][K] bf16 bits
    const unsigned*       __restrict__ Wp,     // packed weights, node offset applied
    const float*          __restrict__ bg,     // global [Hsz]
    unsigned short*       __restrict__ outL,   // LDS [MT][Hsz]
    int wave, int lane)
{
    const int laneq = lane & 15;
    const int half  = lane >> 4;

    v8f acc[RT][2];
#pragma unroll
    for (int rt = 0; rt < RT; ++rt)
#pragma unroll
        for (int ct = 0; ct < 2; ++ct)
            acc[rt][ct] = v8f{0.f,0.f,0.f,0.f,0.f,0.f,0.f,0.f};

    // Per-lane base pointers (k2-contiguous layout)
    const unsigned* wbase0 = Wp + (wave * 32 + 0 * 16 + laneq) * (K / 2) + 8 * half;
    const unsigned* wbase1 = Wp + (wave * 32 + 1 * 16 + laneq) * (K / 2) + 8 * half;

    for (int ks = 0; ks < K / 32; ++ks) {
        const int k0 = ks * 32;

        Frag16 bfr[2];
        bfr[0].q[0] = *(const uint4*)(wbase0 + 16 * ks + 0);
        bfr[0].q[1] = *(const uint4*)(wbase0 + 16 * ks + 4);
        bfr[1].q[0] = *(const uint4*)(wbase1 + 16 * ks + 0);
        bfr[1].q[1] = *(const uint4*)(wbase1 + 16 * ks + 4);

#pragma unroll
        for (int rt = 0; rt < RT; ++rt) {
            const int m = rt * 16 + laneq;
            Frag16 afr;
            // A fragment: u[0..3] = 16B @ (k0 + 16*half), u[4..7] = 16B @ (k0 + 16*half + 8)
            afr.q[0] = *(const uint4*)(actL + m * K + k0 + 16 * half);
            afr.q[1] = *(const uint4*)(actL + m * K + k0 + 16 * half + 8);

            acc[rt][0] = __builtin_amdgcn_wmma_f32_16x16x32_bf16(
                false, afr.v, false, bfr[0].v, (short)0, acc[rt][0], false, false);
            acc[rt][1] = __builtin_amdgcn_wmma_f32_16x16x32_bf16(
                false, afr.v, false, bfr[1].v, (short)0, acc[rt][1], false, false);
        }
    }

    // Epilogue: bias + relu, store bf16. D layout: VGPR v -> M = v + 8*half, lane%16 -> N.
#pragma unroll
    for (int rt = 0; rt < RT; ++rt) {
#pragma unroll
        for (int ct = 0; ct < 2; ++ct) {
            const int col = wave * 32 + ct * 16 + laneq;
            const float bv = bg[col];
#pragma unroll
            for (int v = 0; v < 8; ++v) {
                const int m = rt * 16 + v + 8 * half;
                float x = acc[rt][ct][v] + bv;
                x = x > 0.f ? x : 0.f;
                outL[m * Hsz + col] = f2bf(x);
            }
        }
    }
}

__global__ void __launch_bounds__(256) zero_kernel(float* p, int n) {
    int i = blockIdx.x * 256 + threadIdx.x;
    if (i < n) p[i] = 0.f;
}

__global__ void __launch_bounds__(256) mlp_kernel(
    const float* __restrict__ X, const float* __restrict__ latents,
    const float* __restrict__ constants, const float* __restrict__ scales,
    const float* __restrict__ rotations, const float* __restrict__ centers,
    const unsigned* __restrict__ W0p, const float* __restrict__ b0,
    const unsigned* __restrict__ W1p, const float* __restrict__ b1,
    const unsigned* __restrict__ W2p, const float* __restrict__ b2,
    const float* __restrict__ Wout, const float* __restrict__ bout,
    const int* __restrict__ use_constants, float* __restrict__ acc_out)
{
    // Exactly 64 KB static LDS: two 32 KB ping-pong buffers.
    // bufY front half doubles as the [MT][Esz] layer-0 input; bufY is also
    // reused (as f32) for the output-head reduction scratch at the end.
    __shared__ alignas(16) unsigned short bufX[MT * Hsz];   // 32 KB
    __shared__ alignas(16) unsigned short bufY[MT * Hsz];   // 32 KB

    const int tid  = threadIdx.x;
    const int wave = tid >> 5;
    const int lane = tid & 31;
    const int bid  = blockIdx.x;
    const int mt   = bid & 15;           // point tile (16 tiles of 64)
    const int n    = (bid >> 4) & 31;    // node
    const int b    = bid >> 9;           // batch
    const int p0   = mt * MT;

    // ---- stage inputs: local coords + RBF weight (one thread per point) ----
    // RBF weight lives in a register of thread tid (<MT); same thread consumes it.
    float wreg = 0.f;
    if (tid < MT) {
        const int i = tid, p = p0 + i;
        const float d0 = X[(b * Psz + p) * 3 + 0] - centers[(b * Nsz + n) * 3 + 0];
        const float d1 = X[(b * Psz + p) * 3 + 1] - centers[(b * Nsz + n) * 3 + 1];
        const float d2 = X[(b * Psz + p) * 3 + 2] - centers[(b * Nsz + n) * 3 + 2];
        const float* R = rotations + (b * Nsz + n) * 9;
        const float* s = scales + (b * Nsz + n) * 3;
        const float l0 = (R[0] * d0 + R[1] * d1 + R[2] * d2) / s[0];
        const float l1 = (R[3] * d0 + R[4] * d1 + R[5] * d2) / s[1];
        const float l2 = (R[6] * d0 + R[7] * d1 + R[8] * d2) / s[2];
        bufY[i * Esz + 0] = f2bf(l0);
        bufY[i * Esz + 1] = f2bf(l1);
        bufY[i * Esz + 2] = f2bf(l2);
        wreg = __expf(-0.5f * (l0 * l0 + l1 * l1 + l2 * l2));
        if (use_constants[0]) wreg *= constants[b * Nsz + n];
    }
    // latent broadcast into columns 3..127 (same latent for all points of (b,n))
    for (int idx = tid; idx < MT * Lsz; idx += 256) {
        const int i = idx / Lsz, j = idx - i * Lsz;
        bufY[i * Esz + 3 + j] = f2bf(latents[(n * Bsz + b) * Lsz + j]);
    }
    __syncthreads();

    // ---- 3-layer MLP, all WMMA (ping-pong bufY -> bufX -> bufY -> bufX) ----
    gemm_layer<Esz, MT / 16>(bufY, W0p + n * Hsz * (Esz / 2), b0 + n * Hsz, bufX, wave, lane);
    __syncthreads();
    gemm_layer<Hsz, MT / 16>(bufX, W1p + n * Hsz * (Hsz / 2), b1 + n * Hsz, bufY, wave, lane);
    __syncthreads();
    gemm_layer<Hsz, MT / 16>(bufY, W2p + n * Hsz * (Hsz / 2), b2 + n * Hsz, bufX, wave, lane);
    __syncthreads();

    // ---- output head: 256 -> 1 GEMV, 4 threads per point; bufY reused as f32 scratch ----
    float* red = (float*)bufY;   // [MT][4]
    {
        const int i = tid >> 2;
        const int part = tid & 3;
        const float* wo = Wout + n * Hsz;
        float s = 0.f;
        const int kb = part * 64;
#pragma unroll 8
        for (int k = 0; k < 64; ++k)
            s += bf2f(bufX[i * Hsz + kb + k]) * wo[kb + k];
        red[i * 4 + part] = s;
    }
    __syncthreads();
    if (tid < MT) {
        const int i = tid;
        const float s = red[i * 4 + 0] + red[i * 4 + 1] + red[i * 4 + 2] + red[i * 4 + 3];
        const float pred = s + bout[n] + 0.5f;   // OUTPUT_BIAS
        atomicAdd(&acc_out[b * Psz + p0 + i], wreg * pred);
    }
}

__global__ void __launch_bounds__(256) finalize_kernel(
    const float* __restrict__ acc, float* __restrict__ out)
{
    const int i = blockIdx.x * 256 + threadIdx.x;
    if (i < Bsz * Psz) {
        const float v = acc[i];
        out[i] = v;                                                      // ldif_values
        out[Bsz * Psz + i] = 1.f / (1.f + __expf(100.f * (v + 0.07f))); // sigmoid(100*(-0.07-v))
    }
}

extern "C" void kernel_launch(void* const* d_in, const int* in_sizes, int n_in,
                              void* d_out, int out_size, void* d_ws, size_t ws_size,
                              hipStream_t stream) {
    const float* X          = (const float*)d_in[0];
    const float* latents    = (const float*)d_in[1];
    const float* constants  = (const float*)d_in[2];
    const float* scales     = (const float*)d_in[3];
    const float* rotations  = (const float*)d_in[4];
    const float* centers    = (const float*)d_in[5];
    const float* W0         = (const float*)d_in[6];
    const float* b0         = (const float*)d_in[7];
    const float* W1         = (const float*)d_in[8];
    const float* b1         = (const float*)d_in[9];
    const float* W2         = (const float*)d_in[10];
    const float* b2         = (const float*)d_in[11];
    const float* Wout       = (const float*)d_in[12];
    const float* bout       = (const float*)d_in[13];
    const int*   use_consts = (const int*)d_in[14];

    // workspace layout: [acc: 8192 f32][W0p: 512K dw][W1p: 1M dw][W2p: 1M dw] ~10.5 MB
    float*    acc = (float*)d_ws;
    float*    out = (float*)d_out;
    unsigned* W0p = (unsigned*)((char*)d_ws + Bsz * Psz * sizeof(float));
    unsigned* W1p = W0p + Nsz * Hsz * (Esz / 2);
    unsigned* W2p = W1p + Nsz * Hsz * (Hsz / 2);

    const int tot0 = Nsz * Hsz * (Esz / 2);
    const int tot1 = Nsz * Hsz * (Hsz / 2);
    prepack_kernel<<<(tot0 + 255) / 256, 256, 0, stream>>>(W0, W0p, Esz / 2, tot0);
    prepack_kernel<<<(tot1 + 255) / 256, 256, 0, stream>>>(W1, W1p, Hsz / 2, tot1);
    prepack_kernel<<<(tot1 + 255) / 256, 256, 0, stream>>>(W2, W2p, Hsz / 2, tot1);

    const int npts = Bsz * Psz;
    zero_kernel<<<(npts + 255) / 256, 256, 0, stream>>>(acc, npts);

    const int nblocks = Bsz * Nsz * (Psz / MT);   // 4096
    mlp_kernel<<<nblocks, 256, 0, stream>>>(
        X, latents, constants, scales, rotations, centers,
        W0p, b0, W1p, b1, W2p, b2, Wout, bout, use_consts, acc);

    finalize_kernel<<<(npts + 255) / 256, 256, 0, stream>>>(acc, out);
}
